// ScaledDotProductAttention_21809843929975
// MI455X (gfx1250) — matrix operand order repaired and checked by reading, not compiled, not run on hardware
//
#include <hip/hip_runtime.h>
#include <hip/hip_bf16.h>

// ScaledDotProductAttention  B=4,H=16,S=2048,D=64  (fp32 in, fp32 out).
// Outputs concatenated: context [B,H,S,D] then attn [B,H,S,S].
//
// MI455X (gfx1250, wave32) design:
//  - Store-bandwidth bound: attn = 1.07 GB ≈ 46 us @ 23.3 TB/s. Everything
//    else (68.7 GFLOP of GEMM, exps) must stay under that floor.
//  - bf16 WMMA (v_wmma_f32_16x16x32_bf16) for QK^T and P@V, fp32 accum.
//  - One wave owns a 16-row query tile; two passes over K:
//      pass 1 (maskless, branchless): per-lane anchored exp2 sum + running
//              fmax; single (m,l) butterfly merge after the loop.
//      pass 2: recompute scores, apply mask, p = exp2(s-a)*coef, stage in
//              LDS, coalesced non-temporal b128 attn stores, P@V WMMA.

typedef __attribute__((ext_vector_type(16))) __bf16 v16bf;
typedef __attribute__((ext_vector_type(8)))  float  v8f;
typedef __attribute__((ext_vector_type(4)))  float  v4f;

#define S_LEN 2048
#define D_LEN 64
#define BH    64
#define WAVES 8
#define LDS_STRIDE 33      // 32 cols + 1 pad

#if __has_builtin(__builtin_amdgcn_exp2f)
#define EXP2F(x) __builtin_amdgcn_exp2f(x)
#else
#define EXP2F(x) exp2f(x)
#endif

#define NEG_BIG (-1.0e9f)   // any value with exp2() == 0.0f works here

// 16 contiguous f32 -> v16bf (4 x b128 loads + packed converts)
__device__ __forceinline__ v16bf cvt16(const float* __restrict__ p) {
  const v4f* vp = (const v4f*)p;
  v4f x0 = vp[0], x1 = vp[1], x2 = vp[2], x3 = vp[3];
  v16bf b;
  #pragma unroll
  for (int i = 0; i < 4; ++i) {
    b[i]      = (__bf16)x0[i];
    b[4 + i]  = (__bf16)x1[i];
    b[8 + i]  = (__bf16)x2[i];
    b[12 + i] = (__bf16)x3[i];
  }
  return b;
}

__global__ __launch_bounds__(256)
void sdpa_fused_kernel(const float* __restrict__ Q,
                       const float* __restrict__ K,
                       const float* __restrict__ V,
                       const unsigned char* __restrict__ mask,
                       float* __restrict__ ctx_out,   // [BH,S,D]
                       float* __restrict__ attn_out)  // [BH,S,S]
{
  __shared__ float lds_p[WAVES][16 * LDS_STRIDE];

  const int lane = threadIdx.x & 31;
  const int wave = threadIdx.x >> 5;
  const int bh   = blockIdx.x >> 4;
  const int qgrp = blockIdx.x & 15;
  const int row0 = (qgrp * WAVES + wave) * 16;

  const int m  = lane & 15;     // A-row / B-column within tile
  const int h  = lane >> 4;     // half-wave
  const int kg = h * 16;        // B fragment K-group base

  // log2-domain scale: scores' = (q.k)/sqrt(d) * log2(e)
  const float scale = 0.125f * 1.4426950408889634f;

  const float* Qb = Q + (size_t)bh * S_LEN * D_LEN;
  const float* Kb = K + (size_t)bh * S_LEN * D_LEN;
  const float* Vb = V + (size_t)bh * S_LEN * D_LEN;
  const unsigned char* Mb = mask + (size_t)bh * S_LEN * S_LEN;
  float* ctx  = ctx_out  + (size_t)bh * S_LEN * D_LEN;
  float* attn = attn_out + (size_t)bh * S_LEN * S_LEN;
  float* myP  = &lds_p[wave][0];

  // ---- Q A-fragments (16-bit A 16x32 layout), pre-scaled ----
  const float* qrow = Qb + (size_t)(row0 + m) * D_LEN;
  v16bf qa0, qa1;
  {
    const v4f* qv = (const v4f*)qrow;
    v4f s0 = qv[h * 4],       s1 = qv[h * 4 + 1];
    v4f s2 = qv[h * 4 + 2],   s3 = qv[h * 4 + 3];
    v4f s4 = qv[8 + h * 4],   s5 = qv[8 + h * 4 + 1];
    v4f s6 = qv[8 + h * 4 + 2],  s7 = qv[8 + h * 4 + 3];
    #pragma unroll
    for (int i = 0; i < 4; ++i) {
      qa0[i]      = (__bf16)(s0[i] * scale);
      qa0[4 + i]  = (__bf16)(s1[i] * scale);
      qa0[8 + i]  = (__bf16)(s2[i] * scale);
      qa0[12 + i] = (__bf16)(s3[i] * scale);
      qa1[i]      = (__bf16)(s4[i] * scale);
      qa1[4 + i]  = (__bf16)(s5[i] * scale);
      qa1[8 + i]  = (__bf16)(s6[i] * scale);
      qa1[12 + i] = (__bf16)(s7[i] * scale);
    }
  }

  // ---- pass 1: maskless, branchless per-lane stats ----
  // Anchor a = tile-0 score; per element: pm = fmax(pm,s); pl += exp2(s-a).
  // (Mask not needed: exp2(-1e9 - anything) == 0.0f and masked values never
  //  win the max for real-valued scores, so stats are unchanged.)
  float pm[8], pl[8], anc[8];
  {
    const float* krow = Kb + (size_t)m * D_LEN;   // nt = 0
    v16bf kb0 = cvt16(krow + kg);
    v16bf kb1 = cvt16(krow + 32 + kg);
    v8f c = {};
    c = __builtin_amdgcn_wmma_f32_16x16x32_bf16(false, qa0, false, kb0,
                                                (short)0, c, false, false);
    c = __builtin_amdgcn_wmma_f32_16x16x32_bf16(false, qa1, false, kb1,
                                                (short)0, c, false, false);
    #pragma unroll
    for (int r = 0; r < 8; ++r) { anc[r] = c[r]; pm[r] = c[r]; pl[r] = 1.0f; }
  }

  for (int nt = 1; nt < S_LEN / 16; ++nt) {
    const float* krow = Kb + (size_t)(nt * 16 + m) * D_LEN;
    if (nt + 1 < S_LEN / 16)
      __builtin_prefetch(Kb + (size_t)((nt + 1) * 16 + m) * D_LEN, 0, 1);

    v16bf kb0 = cvt16(krow + kg);
    v16bf kb1 = cvt16(krow + 32 + kg);

    v8f c = {};
    c = __builtin_amdgcn_wmma_f32_16x16x32_bf16(false, qa0, false, kb0,
                                                (short)0, c, false, false);
    c = __builtin_amdgcn_wmma_f32_16x16x32_bf16(false, qa1, false, kb1,
                                                (short)0, c, false, false);

    #pragma unroll
    for (int r = 0; r < 8; ++r) {
      const float s = c[r];
      pm[r] = fmaxf(pm[r], s);
      pl[r] += EXP2F(s - anc[r]);        // 1 exp2 + fadd, no branch
    }
  }

  // ---- merge (m,l) across the 16-lane group; fold into per-row coef ----
  float coef[8];
  #pragma unroll
  for (int r = 0; r < 8; ++r) {
    float mv = pm[r];
    float lv = pl[r] * EXP2F(anc[r] - pm[r]);   // rebase to own max
    #pragma unroll
    for (int off = 8; off > 0; off >>= 1) {
      float mo = __shfl_xor(mv, off, 32);
      float lo = __shfl_xor(lv, off, 32);
      float nm = fmaxf(mv, mo);
      lv = lv * EXP2F(mv - nm) + lo * EXP2F(mo - nm);
      mv = nm;
    }
    // p = exp2(s - anc) * coef, coef = exp2(anc - m_final) / l_total
    coef[r] = EXP2F(anc[r] - mv) * (1.0f / lv);
  }

  // ---- pass 2: recompute, mask, NT-write attn, P@V via WMMA ----
  v8f acc[4];
  #pragma unroll
  for (int t = 0; t < 4; ++t) acc[t] = (v8f){};

  for (int ch = 0; ch < S_LEN / 32; ++ch) {
    #pragma unroll
    for (int sub = 0; sub < 2; ++sub) {
      const int nt = ch * 2 + sub;
      const float* krow = Kb + (size_t)(nt * 16 + m) * D_LEN;
      v16bf kb0 = cvt16(krow + kg);
      v16bf kb1 = cvt16(krow + 32 + kg);

      v8f c = {};
      c = __builtin_amdgcn_wmma_f32_16x16x32_bf16(false, qa0, false, kb0,
                                                  (short)0, c, false, false);
      c = __builtin_amdgcn_wmma_f32_16x16x32_bf16(false, qa1, false, kb1,
                                                  (short)0, c, false, false);

      #pragma unroll
      for (int r = 0; r < 8; ++r) {
        const int mloc = h * 8 + r;
        const int mrow = row0 + mloc;
        const int col  = nt * 16 + m;
        const bool msk = Mb[(size_t)mrow * S_LEN + col] != 0;
        const float e  = msk ? NEG_BIG : (c[r] - anc[r]);
        const float p  = EXP2F(e) * coef[r];
        myP[mloc * LDS_STRIDE + sub * 16 + m] = p;
      }
    }
    __builtin_amdgcn_wave_barrier();   // order DS stores before DS loads

    // coalesced non-temporal b128 stores of the 16x32 probability tile
    {
      const size_t colbase = (size_t)ch * 32;
      #pragma unroll
      for (int rr = 0; rr < 4; ++rr) {
        const int row = rr * 4 + (lane >> 3);
        const int c4  = (lane & 7) * 4;
        v4f pv;
        #pragma unroll
        for (int i = 0; i < 4; ++i)
          pv[i] = myP[row * LDS_STRIDE + c4 + i];
        __builtin_nontemporal_store(
            pv, (v4f*)(attn + (size_t)(row0 + row) * S_LEN + colbase + c4));
      }
    }

    // P as A-fragment (16x32 bf16), read back in A layout
    v16bf pa;
    #pragma unroll
    for (int e = 0; e < 8; ++e) {
      pa[e]     = (__bf16)myP[m * LDS_STRIDE + h * 16 + e];
      pa[8 + e] = (__bf16)myP[m * LDS_STRIDE + h * 16 + 8 + e];
    }
    __builtin_amdgcn_wave_barrier();   // WAR vs next chunk's stores

    // V as B-fragments: lane holds d-column m, k = ch*32 + kg + e
    const float* vbase = Vb + ((size_t)(ch * 32 + kg)) * D_LEN + m;
    #pragma unroll
    for (int t = 0; t < 4; ++t) {
      v16bf vb;
      #pragma unroll
      for (int e = 0; e < 16; ++e)
        vb[e] = (__bf16)vbase[(size_t)e * D_LEN + t * 16];
      acc[t] = __builtin_amdgcn_wmma_f32_16x16x32_bf16(false, pa, false, vb,
                                                       (short)0, acc[t],
                                                       false, false);
    }
  }

  // ---- context out: acc[t][r] -> (row0 + h*8 + r, t*16 + m) ----
  #pragma unroll
  for (int t = 0; t < 4; ++t) {
    #pragma unroll
    for (int r = 0; r < 8; ++r) {
      const int mrow = row0 + h * 8 + r;
      __builtin_nontemporal_store(
          acc[t][r], ctx + (size_t)mrow * D_LEN + t * 16 + m);
    }
  }
}

extern "C" void kernel_launch(void* const* d_in, const int* in_sizes, int n_in,
                              void* d_out, int out_size, void* d_ws, size_t ws_size,
                              hipStream_t stream) {
  (void)in_sizes; (void)n_in; (void)out_size; (void)d_ws; (void)ws_size;
  const float* Q = (const float*)d_in[0];
  const float* K = (const float*)d_in[1];
  const float* V = (const float*)d_in[2];
  const unsigned char* mask = (const unsigned char*)d_in[3];

  float* out  = (float*)d_out;
  float* ctx  = out;                               // context first
  float* attn = out + (size_t)BH * S_LEN * D_LEN;  // then attn

  dim3 grid(BH * (S_LEN / (16 * WAVES)));
  dim3 block(32 * WAVES);
  sdpa_fused_kernel<<<grid, block, 0, stream>>>(Q, K, V, mask, ctx, attn);
}
